// AALSTM_53566832116443
// MI455X (gfx1250) — compile-verified, matrix operand order checked by reading, not run on hardware
//
#include <hip/hip_runtime.h>
#include <cstdint>

typedef __attribute__((ext_vector_type(16))) __bf16      v16bf;
typedef __attribute__((ext_vector_type(8)))  float       v8f;
typedef __attribute__((ext_vector_type(4)))  unsigned int v4u;

union FragU { v16bf bf; v4u u[2]; };

// ---------------- ws layout (bytes, all 256-aligned) ----------------
static constexpr size_t OFF_STATS = 0;         // mean[32], rstd[32] f32
static constexpr size_t OFF_BSUM1 = 256;       // 2048 f32  (b_ih1 + b_hh1)
static constexpr size_t OFF_BSUM2 = 8448;      // 1024 f32
static constexpr size_t OFF_WIH1  = 12544;     // 128 tiles  * 1KB  (2048x32)
static constexpr size_t OFF_WHH1  = 143616;    // 2048 tiles * 1KB  (2048x512)
static constexpr size_t OFF_WIH2  = 2240768;   // 1024 tiles * 1KB  (1024x512)
static constexpr size_t OFF_WHH2  = 3289344;   // 512 tiles  * 1KB  (1024x256)
static constexpr size_t OFF_XN    = 3813632;   // [1024][128][32] bf16
static constexpr size_t OFF_H0    = 12202240;  // 2 * [128][512] bf16 (double buffered)
static constexpr size_t OFF_H1    = 12464384;  // 2 * [128][256] bf16
static constexpr size_t OFF_BAR   = 12595456;  // 2 u32 barrier state

#define NUM_WG 48
// LDS layout: cell1 WGs: [4][17] B tiles (68KB) + inp[128][32] bf16 (8KB)
//             cell2 WGs: [4][24] B tiles (96KB)
#define LDS_BYTES   98304
#define LDS_INP_OFF 69632   // 4*17*1024

// ---------------- helpers ----------------
__device__ __forceinline__ unsigned short f2bf(float x) {
  unsigned u = __float_as_uint(x);
  unsigned r = (u + 0x7FFFu + ((u >> 16) & 1u)) >> 16;  // round-to-nearest-even
  return (unsigned short)r;
}
__device__ __forceinline__ float bf2f(unsigned short h) {
  return __uint_as_float(((unsigned)h) << 16);
}
__device__ __forceinline__ float sigm(float x) { return 1.0f / (1.0f + __expf(-x)); }

// A fragment: 16x32 bf16 tile from a row-major [M][ld] bf16 array (global or LDS).
// lane l: row = m0 + (l&15); K chunks at k0+kh and k0+16+kh, kh = 8*(l>=16).
__device__ __forceinline__ v16bf load_a(const unsigned short* base,
                                        int ld, int m0, int k0, int lane) {
  const int kh = (lane >> 4) << 3;
  const unsigned short* p = base + (size_t)(m0 + (lane & 15)) * ld + k0 + kh;
  FragU f;
  f.u[0] = *(const v4u*)p;
  f.u[1] = *(const v4u*)(p + 16);
  return f.bf;
}

// B fragment: pre-packed 1KB tile, lane-major (lane*16 ushorts each).
__device__ __forceinline__ v16bf load_b(const unsigned short* tile, int lane) {
  const v4u* p = (const v4u*)(tile + (size_t)lane * 16);
  FragU f;
  f.u[0] = p[0];
  f.u[1] = p[1];
  return f.bf;
}

__device__ __forceinline__ v8f wmma_bf16(v16bf a, v16bf b, v8f c) {
  return __builtin_amdgcn_wmma_f32_16x16x32_bf16(false, a, false, b, (short)0, c, false, false);
}

__device__ __forceinline__ v8f zero8() {
  v8f z = {0.f, 0.f, 0.f, 0.f, 0.f, 0.f, 0.f, 0.f};
  return z;
}

// Global barrier: counter + monotonically increasing generation.
__device__ __forceinline__ void grid_sync(unsigned* bar, unsigned target) {
  __syncthreads();
  if (threadIdx.x == 0) {
    __threadfence();
    unsigned prev = atomicAdd(&bar[0], 1u);
    if (prev == NUM_WG - 1) {
      __hip_atomic_store(&bar[0], 0u, __ATOMIC_RELAXED, __HIP_MEMORY_SCOPE_AGENT);
      __hip_atomic_fetch_add(&bar[1], 1u, __ATOMIC_RELEASE, __HIP_MEMORY_SCOPE_AGENT);
    } else {
      while (__hip_atomic_load(&bar[1], __ATOMIC_ACQUIRE, __HIP_MEMORY_SCOPE_AGENT) < target) {
        __builtin_amdgcn_s_sleep(2);
      }
    }
  }
  __syncthreads();
}

// ---------------- BatchNorm stats: one block per feature ----------------
__global__ void k_bnstats(const float* __restrict__ x, float* __restrict__ stats) {
  const int f = blockIdx.x;  // 0..31
  float s = 0.f, sq = 0.f;
  for (int i = threadIdx.x; i < 128 * 1024; i += 256) {
    float v = x[(size_t)i * 32 + f];
    s += v;
    sq += v * v;
  }
  __shared__ float sh1[256], sh2[256];
  sh1[threadIdx.x] = s;
  sh2[threadIdx.x] = sq;
  __syncthreads();
  for (int off = 128; off > 0; off >>= 1) {
    if (threadIdx.x < off) {
      sh1[threadIdx.x] += sh1[threadIdx.x + off];
      sh2[threadIdx.x] += sh2[threadIdx.x + off];
    }
    __syncthreads();
  }
  if (threadIdx.x == 0) {
    float mean = sh1[0] * (1.0f / 131072.0f);
    float var  = sh2[0] * (1.0f / 131072.0f) - mean * mean;
    stats[f]      = mean;
    stats[32 + f] = rsqrtf(var + 1e-5f);
  }
}

// ---------------- normalize + convert to bf16, relayout [B,T,F] -> [T,B,F] ----------------
__global__ void k_bnnorm(const float* __restrict__ x, const float* __restrict__ stats,
                         const float* __restrict__ gamma, const float* __restrict__ beta,
                         unsigned short* __restrict__ xnbf) {
  size_t o = (size_t)blockIdx.x * 256 + threadIdx.x;  // over [t][b][f], 4194304 total
  int f = (int)(o & 31);
  size_t bt = o >> 5;
  int b = (int)(bt & 127);
  int t = (int)(bt >> 7);
  float v  = x[((size_t)b * 1024 + t) * 32 + f];
  float xn = (v - stats[f]) * stats[32 + f] * gamma[f] + beta[f];
  xnbf[o] = f2bf(xn);
}

// ---------------- pack W [N x K] f32 row-major -> WMMA B-fragment tiles ----------------
__global__ void k_pack(const float* __restrict__ W, unsigned short* __restrict__ dst,
                       int N, int K) {
  int t = blockIdx.x * 256 + threadIdx.x;
  int e    = t & 15;
  int ln   = (t >> 4) & 31;
  int tile = t >> 9;
  int KT = K >> 5;
  int kt = tile % KT;
  int nt = tile / KT;
  if (nt >= (N >> 4)) return;
  int n = nt * 16 + (ln & 15);
  int k = kt * 32 + ((e >> 3) << 4) + ((ln >> 4) << 3) + (e & 7);
  dst[t] = f2bf(W[(size_t)n * K + k]);
}

// ---------------- bias sums, zero h buffers, reset barrier ----------------
__global__ void k_prep(const float* __restrict__ bih1, const float* __restrict__ bhh1,
                       const float* __restrict__ bih2, const float* __restrict__ bhh2,
                       float* __restrict__ bsum1, float* __restrict__ bsum2,
                       unsigned short* __restrict__ h0bf, unsigned short* __restrict__ h1bf,
                       unsigned* __restrict__ bar) {
  int t = blockIdx.x * 256 + threadIdx.x;
  if (t < 2048) {
    bsum1[t] = bih1[t] + bhh1[t];
  } else if (t < 3072) {
    int j = t - 2048;
    bsum2[j] = bih2[j] + bhh2[j];
  } else if (t < 134144) {
    h0bf[t - 3072] = 0;
  } else if (t < 199680) {
    h1bf[t - 134144] = 0;
  }
  if (t == 0) { bar[0] = 0u; bar[1] = 0u; }
}

// ---------------- persistent LSTM decode kernel ----------------
__global__ __launch_bounds__(256, 1) void k_lstm(
    const unsigned short* __restrict__ wih1p, const unsigned short* __restrict__ whh1p,
    const unsigned short* __restrict__ wih2p, const unsigned short* __restrict__ whh2p,
    const float* __restrict__ bsum1, const float* __restrict__ bsum2,
    const unsigned short* __restrict__ xnbf,
    unsigned short* __restrict__ h0bf,  // 2 buffers of 128*512
    unsigned short* __restrict__ h1bf,  // 2 buffers of 128*256
    const float* __restrict__ wfc, const float* __restrict__ bfc,
    float* __restrict__ out,            // [128][1024]
    unsigned* __restrict__ bar) {
  extern __shared__ __align__(16) char smem[];
  unsigned short* ldsB   = (unsigned short*)smem;                  // B-fragment tiles
  unsigned short* ldsInp = (unsigned short*)(smem + LDS_INP_OFF);  // [128][32] (cell1 WGs)

  const int tid  = threadIdx.x;
  const int lane = tid & 31;
  const int wave = tid >> 5;
  const int wg   = blockIdx.x;
  const bool isA = (wg < 32);

  const int mt = wave;                   // both roles: wave owns M-tile
  const int nc = isA ? wg : (wg - 32);   // column bundle (16 hidden cols)
  const int colsub = lane & 15;
  const int khalf  = (lane >> 4) << 3;
  const int col    = nc * 16 + colsub;
  const int mbase  = mt * 16 + khalf;    // + r = global M row of acc element r

  float bI, bF, bG, bO;
  if (isA) {
    bI = bsum1[col]; bF = bsum1[512 + col]; bG = bsum1[1024 + col]; bO = bsum1[1536 + col];
  } else {
    bI = bsum2[col]; bF = bsum2[256 + col]; bG = bsum2[512 + col]; bO = bsum2[768 + col];
  }
  float cst[8];
#pragma unroll
  for (int r = 0; r < 8; ++r) cst[r] = 0.0f;
  const float bfc0 = bfc[0];

  // ---- prologue: cache this WG's step-invariant B fragments in LDS ----
  if (isA) {
    // tiles [g][kk]: kk=0 -> Wih1 tile; kk=1..16 -> Whh1 k-tile kk-1
    for (int i = tid; i < 4 * 17 * 64; i += 256) {
      int tile = i >> 6, ch = i & 63;
      int g = tile / 17, kk = tile % 17;
      const unsigned short* gsrc =
          (kk == 0) ? (wih1p + (size_t)(g * 32 + nc) * 512)
                    : (whh1p + ((size_t)(g * 32 + nc) * 16 + (kk - 1)) * 512);
      ((v4u*)ldsB)[i] = ((const v4u*)gsrc)[ch];
    }
    // inp(0) = xn[t=0] slice ([128][32] bf16)
    for (int i = tid; i < 4096; i += 256) ldsInp[i] = xnbf[i];
  } else {
    // tiles [g][kk]: kk=0..15 -> Wih2 k-tile; kk=16..23 -> Whh2 k-tile kk-16
    for (int i = tid; i < 4 * 24 * 64; i += 256) {
      int tile = i >> 6, ch = i & 63;
      int g = tile / 24, kk = tile % 24;
      const unsigned short* gsrc =
          (kk < 16) ? (wih2p + ((size_t)(g * 16 + nc) * 16 + kk) * 512)
                    : (whh2p + ((size_t)(g * 16 + nc) * 8 + (kk - 16)) * 512);
      ((v4u*)ldsB)[i] = ((const v4u*)gsrc)[ch];
    }
  }
  __syncthreads();

  unsigned target = 1;
  for (int t = 0; t < 1024; ++t) {
    const int rb = t & 1, wb = (t + 1) & 1;
    const unsigned short* h0r = h0bf + (size_t)rb * (128 * 512);
    unsigned short*       h0w = h0bf + (size_t)wb * (128 * 512);
    const unsigned short* h1r = h1bf + (size_t)rb * (128 * 256);
    unsigned short*       h1w = h1bf + (size_t)wb * (128 * 256);

    // ---- Phase A: cell1 gates (inp @ Wih1^T + h0 @ Whh1^T) + LSTM update ----
    if (isA) {
      v8f a0 = zero8(), a1 = zero8(), a2 = zero8(), a3 = zero8();
      {  // kk = 0: input features (A and B both from LDS)
        v16bf af = load_a(ldsInp, 32, mt * 16, 0, lane);
        v16bf b0 = load_b(ldsB + (size_t)(0 * 17 + 0) * 512, lane);
        v16bf b1 = load_b(ldsB + (size_t)(1 * 17 + 0) * 512, lane);
        v16bf b2 = load_b(ldsB + (size_t)(2 * 17 + 0) * 512, lane);
        v16bf b3 = load_b(ldsB + (size_t)(3 * 17 + 0) * 512, lane);
        a0 = wmma_bf16(af, b0, a0);
        a1 = wmma_bf16(af, b1, a1);
        a2 = wmma_bf16(af, b2, a2);
        a3 = wmma_bf16(af, b3, a3);
      }
#pragma unroll 4
      for (int kt = 0; kt < 16; ++kt) {  // K-tiles over h0 (512), A from global
        v16bf af = load_a(h0r, 512, mt * 16, kt * 32, lane);
        v16bf b0 = load_b(ldsB + (size_t)(0 * 17 + 1 + kt) * 512, lane);
        v16bf b1 = load_b(ldsB + (size_t)(1 * 17 + 1 + kt) * 512, lane);
        v16bf b2 = load_b(ldsB + (size_t)(2 * 17 + 1 + kt) * 512, lane);
        v16bf b3 = load_b(ldsB + (size_t)(3 * 17 + 1 + kt) * 512, lane);
        a0 = wmma_bf16(af, b0, a0);
        a1 = wmma_bf16(af, b1, a1);
        a2 = wmma_bf16(af, b2, a2);
        a3 = wmma_bf16(af, b3, a3);
      }
#pragma unroll
      for (int r = 0; r < 8; ++r) {
        float iv = sigm(a0[r] + bI);
        float fv = sigm(a1[r] + bF);
        float gv = tanhf(a2[r] + bG);
        float ov = sigm(a3[r] + bO);
        float c  = fv * cst[r] + iv * gv;
        cst[r] = c;
        float h = ov * tanhf(c);
        h0w[(size_t)(mbase + r) * 512 + col] = f2bf(h);
      }
    }
    grid_sync(bar, target++);

    // ---- Phase B: cell2 gates (h0' @ Wih2^T + h1 @ Whh2^T) + LSTM update ----
    if (!isA) {
      v8f a0 = zero8(), a1 = zero8(), a2 = zero8(), a3 = zero8();
#pragma unroll 4
      for (int kt = 0; kt < 16; ++kt) {  // K-tiles over h0' (512)
        v16bf af = load_a(h0w, 512, mt * 16, kt * 32, lane);
        v16bf b0 = load_b(ldsB + (size_t)(0 * 24 + kt) * 512, lane);
        v16bf b1 = load_b(ldsB + (size_t)(1 * 24 + kt) * 512, lane);
        v16bf b2 = load_b(ldsB + (size_t)(2 * 24 + kt) * 512, lane);
        v16bf b3 = load_b(ldsB + (size_t)(3 * 24 + kt) * 512, lane);
        a0 = wmma_bf16(af, b0, a0);
        a1 = wmma_bf16(af, b1, a1);
        a2 = wmma_bf16(af, b2, a2);
        a3 = wmma_bf16(af, b3, a3);
      }
#pragma unroll 4
      for (int kt = 0; kt < 8; ++kt) {   // K-tiles over h1 (256)
        v16bf af = load_a(h1r, 256, mt * 16, kt * 32, lane);
        v16bf b0 = load_b(ldsB + (size_t)(0 * 24 + 16 + kt) * 512, lane);
        v16bf b1 = load_b(ldsB + (size_t)(1 * 24 + 16 + kt) * 512, lane);
        v16bf b2 = load_b(ldsB + (size_t)(2 * 24 + 16 + kt) * 512, lane);
        v16bf b3 = load_b(ldsB + (size_t)(3 * 24 + 16 + kt) * 512, lane);
        a0 = wmma_bf16(af, b0, a0);
        a1 = wmma_bf16(af, b1, a1);
        a2 = wmma_bf16(af, b2, a2);
        a3 = wmma_bf16(af, b3, a3);
      }
#pragma unroll
      for (int r = 0; r < 8; ++r) {
        float iv = sigm(a0[r] + bI);
        float fv = sigm(a1[r] + bF);
        float gv = tanhf(a2[r] + bG);
        float ov = sigm(a3[r] + bO);
        float c  = fv * cst[r] + iv * gv;
        cst[r] = c;
        float h = ov * tanhf(c);
        h1w[(size_t)(mbase + r) * 256 + col] = f2bf(h);
      }
    }
    grid_sync(bar, target++);

    // ---- Phase C (redundant per cell1 WG): out = h1' @ Wfc^T + b; next input ----
    // Only cell1 WGs consume `inp`, so each builds its own LDS copy; no 3rd barrier.
    if (isA) {
      if (tid < 128) {
        const int m = tid;
        float acc = bfc0;
        const unsigned short* hrow = h1w + (size_t)m * 256;
#pragma unroll 8
        for (int c2 = 0; c2 < 256; ++c2) acc += bf2f(hrow[c2]) * wfc[c2];
        if (wg == 0) out[(size_t)m * 1024 + t] = acc;  // single writer of d_out
        if (t + 1 < 1024) {
          const unsigned short* xrow = xnbf + ((size_t)(t + 1) * 128 + m) * 32;
          unsigned short* irow = ldsInp + (size_t)m * 32;
#pragma unroll
          for (int f = 0; f < 31; ++f) irow[f] = xrow[f];
          irow[31] = f2bf(acc);
        }
      }
      __syncthreads();  // inp visible WG-wide before phase A(t+1)
    }
  }
}

extern "C" void kernel_launch(void* const* d_in, const int* in_sizes, int n_in,
                              void* d_out, int out_size, void* d_ws, size_t ws_size,
                              hipStream_t stream) {
  (void)in_sizes; (void)n_in; (void)out_size; (void)ws_size;
  const float* x     = (const float*)d_in[0];
  // d_in[1] = dagger_gt (unused by the reference)
  const float* Wih1  = (const float*)d_in[2];
  const float* Whh1  = (const float*)d_in[3];
  const float* bih1  = (const float*)d_in[4];
  const float* bhh1  = (const float*)d_in[5];
  const float* Wih2  = (const float*)d_in[6];
  const float* Whh2  = (const float*)d_in[7];
  const float* bih2  = (const float*)d_in[8];
  const float* bhh2  = (const float*)d_in[9];
  const float* Wfc   = (const float*)d_in[10];
  const float* bfc   = (const float*)d_in[11];
  const float* gamma = (const float*)d_in[12];
  const float* beta  = (const float*)d_in[13];

  char* ws = (char*)d_ws;
  float*          stats = (float*)(ws + OFF_STATS);
  float*          bsum1 = (float*)(ws + OFF_BSUM1);
  float*          bsum2 = (float*)(ws + OFF_BSUM2);
  unsigned short* wih1p = (unsigned short*)(ws + OFF_WIH1);
  unsigned short* whh1p = (unsigned short*)(ws + OFF_WHH1);
  unsigned short* wih2p = (unsigned short*)(ws + OFF_WIH2);
  unsigned short* whh2p = (unsigned short*)(ws + OFF_WHH2);
  unsigned short* xnbf  = (unsigned short*)(ws + OFF_XN);
  unsigned short* h0bf  = (unsigned short*)(ws + OFF_H0);
  unsigned short* h1bf  = (unsigned short*)(ws + OFF_H1);
  unsigned*       bar   = (unsigned*)(ws + OFF_BAR);

  k_bnstats<<<32, 256, 0, stream>>>(x, stats);
  k_bnnorm<<<16384, 256, 0, stream>>>(x, stats, gamma, beta, xnbf);
  k_pack<<<256, 256, 0, stream>>>(Wih1, wih1p, 2048, 32);
  k_pack<<<4096, 256, 0, stream>>>(Whh1, whh1p, 2048, 512);
  k_pack<<<2048, 256, 0, stream>>>(Wih2, wih2p, 1024, 512);
  k_pack<<<1024, 256, 0, stream>>>(Whh2, whh2p, 1024, 256);
  k_prep<<<780, 256, 0, stream>>>(bih1, bhh1, bih2, bhh2, bsum1, bsum2,
                                  h0bf, h1bf, bar);
  k_lstm<<<NUM_WG, 256, LDS_BYTES, stream>>>(wih1p, whh1p, wih2p, whh2p, bsum1, bsum2,
                                             xnbf, h0bf, h1bf, Wfc, bfc,
                                             (float*)d_out, bar);
}
